// TPALSTM_63539746177214
// MI455X (gfx1250) — compile-verified, matrix operand order checked by reading, not run on hardware
//
#include <hip/hip_runtime.h>
#include <hip/hip_bf16.h>

// ---------------- problem constants ----------------
#define HID     128
#define LAYERS  2
#define HORIZON 12
#define OBS     24
#define FNUM    32
#define BE      20800            // 64 * 325 flattened rows
#define GATES   512              // 4 * HID

typedef __bf16 bf16_t;
typedef __attribute__((ext_vector_type(8)))  __bf16 v8bf;
typedef __attribute__((ext_vector_type(16))) __bf16 v16bf;
typedef __attribute__((ext_vector_type(8)))  float  v8f;

__device__ __forceinline__ v8f zero8f() {
    v8f z;
#pragma unroll
    for (int i = 0; i < 8; ++i) z[i] = 0.0f;
    return z;
}
__device__ __forceinline__ v16bf zero16bf() {
    v16bf z;
#pragma unroll
    for (int i = 0; i < 16; ++i) z[i] = (bf16_t)0.0f;
    return z;
}
// build a 16-element bf16 fragment from two contiguous 16-byte halves
__device__ __forceinline__ v16bf load_frag2(const bf16_t* p0, const bf16_t* p1) {
    v8bf a = *(const v8bf*)p0;
    v8bf b = *(const v8bf*)p1;
    v16bf r;
#pragma unroll
    for (int i = 0; i < 8; ++i) { r[i] = a[i]; r[i + 8] = b[i]; }
    return r;
}
__device__ __forceinline__ v8f wmma_bf16(v16bf a, v16bf b, v8f c) {
    // D = A(16x32 bf16) * B(32x16 bf16) + C(16x16 f32)
    return __builtin_amdgcn_wmma_f32_16x16x32_bf16(
        /*neg_a=*/false, a, /*neg_b=*/false, b,
        /*c_mod=*/(short)0, c, /*reuse_a=*/false, /*reuse_b=*/false);
}
__device__ __forceinline__ float fsig(float x)  { return 1.0f / (1.0f + __expf(-x)); }
__device__ __forceinline__ float ftanh(float x) { return 1.0f - 2.0f / (1.0f + __expf(2.0f * x)); }

// ================= kernel 1: input projection (xc = relu(x*Wh^T + bh)) =================
// writes bf16 buffer laid out (row, t, h) contiguous
__global__ void xc_kernel(const float* __restrict__ x, const float* __restrict__ Wh,
                          const float* __restrict__ bh, bf16_t* __restrict__ X, long total) {
    long i = (long)blockIdx.x * blockDim.x + threadIdx.x;
    if (i >= total) return;
    int  h  = (int)(i & (HID - 1));
    long rt = i >> 7;                       // == r*OBS + t  (x.reshape(-1,T) is a flat reinterpretation)
    float v = x[rt] * Wh[h] + bh[h];
    X[i] = (bf16_t)fmaxf(v, 0.0f);
}

// ================= kernel 2: one LSTM layer, all 24 steps, in-place buf =================
// block = 128 threads (4 waves); each wave owns a 16-row batch tile.
// Dynamic LDS: W_ih frags (64K bf16) | W_hh frags (64K bf16) | bias[512] f32 | stage 4*16x128 bf16
__global__ void __launch_bounds__(128, 1)
lstm_layer_kernel(const float* __restrict__ W_ih, const float* __restrict__ W_hh,
                  const float* __restrict__ b_ih, const float* __restrict__ b_hh,
                  bf16_t* __restrict__ buf) {
    extern __shared__ __align__(32) char smem[];
    bf16_t* Wihs  = (bf16_t*)smem;                 // 65536 bf16  (128 frags * 512)
    bf16_t* Whhs  = Wihs + 65536;                  // 65536 bf16
    float*  biasS = (float*)(Whhs + 65536);        // 512 f32
    bf16_t* stage = (bf16_t*)(biasS + 512);        // 4 * 2048 bf16

    // ---- fill weights into LDS in exact B-fragment order ----
    // B[k][n] = W[n][k]; fragment(nt,kq): lanes 0-15 -> N=lane, K elems {0..7,16..23};
    // lanes 16-31 -> N=lane-16, K elems {8..15,24..31}; 16 bf16 per lane, contiguous.
    for (int idx = threadIdx.x; idx < 65536; idx += blockDim.x) {
        int frag   = idx >> 9;            // nt*4 + kq
        int within = idx & 511;
        int ln     = within >> 4;         // lane 0..31
        int j      = within & 15;         // element 0..15
        int nt = frag >> 2, kq = frag & 3;
        int nn = nt * 16 + (ln & 15);
        int kk = kq * 32 + ((ln < 16) ? 0 : 8) + ((j < 8) ? j : (j - 8) + 16);
        Wihs[idx] = (bf16_t)W_ih[nn * HID + kk];
        Whhs[idx] = (bf16_t)W_hh[nn * HID + kk];
    }
    for (int i = threadIdx.x; i < GATES; i += blockDim.x)
        biasS[i] = b_ih[i] + b_hh[i];
    __syncthreads();

    const int wv    = threadIdx.x >> 5;
    const int lane  = threadIdx.x & 31;
    const int lhalf = lane & 15;
    const int khalf = (lane < 16) ? 0 : 8;   // K (and M) offset for high half-wave
    bf16_t* myStage = stage + wv * 2048;     // 16 x 128 bf16
    const size_t rowbase = ((size_t)blockIdx.x * 4 + wv) * 16;
    const size_t arow    = rowbase + lhalf;  // row this lane loads A-fragments for

    v16bf hA[4];
    float cst[8][8];
#pragma unroll
    for (int q = 0; q < 4; ++q) hA[q] = zero16bf();
#pragma unroll
    for (int a = 0; a < 8; ++a)
#pragma unroll
        for (int b = 0; b < 8; ++b) cst[a][b] = 0.0f;

    for (int t = 0; t < OBS; ++t) {
        // ---- load x_t A-fragments straight from global (already A-layout friendly) ----
        v16bf xA[4];
#pragma unroll
        for (int kq = 0; kq < 4; ++kq) {
            const bf16_t* p = buf + (arow * OBS + (size_t)t) * HID + kq * 32 + khalf;
            xA[kq] = load_frag2(p, p + 16);
        }

        // ---- gates for 8 hidden-column chunks of 16 ----
#pragma unroll
        for (int hc = 0; hc < 8; ++hc) {
            v8f ai = zero8f(), af = zero8f(), ag = zero8f(), ao = zero8f();
            const int fi = (hc)      * 4;   // frag base index (nt*4) for gate i
            const int ff = (8  + hc) * 4;
            const int fg = (16 + hc) * 4;
            const int fo = (24 + hc) * 4;
#pragma unroll
            for (int kq = 0; kq < 4; ++kq) {
                const int lo = lane * 16;
                const bf16_t* pii = Wihs + ((fi + kq) << 9) + lo;
                const bf16_t* pif = Wihs + ((ff + kq) << 9) + lo;
                const bf16_t* pig = Wihs + ((fg + kq) << 9) + lo;
                const bf16_t* pio = Wihs + ((fo + kq) << 9) + lo;
                const bf16_t* phi = Whhs + ((fi + kq) << 9) + lo;
                const bf16_t* phf = Whhs + ((ff + kq) << 9) + lo;
                const bf16_t* phg = Whhs + ((fg + kq) << 9) + lo;
                const bf16_t* pho = Whhs + ((fo + kq) << 9) + lo;
                ai = wmma_bf16(xA[kq], load_frag2(pii, pii + 8), ai);
                ai = wmma_bf16(hA[kq], load_frag2(phi, phi + 8), ai);
                af = wmma_bf16(xA[kq], load_frag2(pif, pif + 8), af);
                af = wmma_bf16(hA[kq], load_frag2(phf, phf + 8), af);
                ag = wmma_bf16(xA[kq], load_frag2(pig, pig + 8), ag);
                ag = wmma_bf16(hA[kq], load_frag2(phg, phg + 8), ag);
                ao = wmma_bf16(xA[kq], load_frag2(pio, pio + 8), ao);
                ao = wmma_bf16(hA[kq], load_frag2(pho, pho + 8), ao);
            }
            // ---- activations + state update (D-tile: VGPR r, lane -> M=r+khalf, N=col) ----
            const int col = hc * 16 + lhalf;
            const float bI = biasS[col], bF = biasS[HID + col];
            const float bG = biasS[2 * HID + col], bO = biasS[3 * HID + col];
#pragma unroll
            for (int r = 0; r < 8; ++r) {
                float iv = fsig(ai[r] + bI);
                float fv = fsig(af[r] + bF);
                float gv = ftanh(ag[r] + bG);
                float ov = fsig(ao[r] + bO);
                float cn = fv * cst[hc][r] + iv * gv;
                cst[hc][r] = cn;
                float hn = ov * ftanh(cn);
                myStage[(r + khalf) * HID + col] = (bf16_t)hn;   // row-major 16x128 tile
            }
        }

        // ---- reload h as A-fragments (lane transpose through LDS; per-wave region, in-order DS) ----
#pragma unroll
        for (int kq = 0; kq < 4; ++kq) {
            const bf16_t* p = myStage + lhalf * HID + kq * 32 + khalf;
            hA[kq] = load_frag2(p, p + 16);
        }

        // ---- coalesced copy stage -> global (this layer's output at time t, in-place) ----
        {
            const int srow = lane >> 1, part = lane & 1;
            const uint4* src = (const uint4*)(myStage + srow * HID + part * 64);
            uint4* dst = (uint4*)(buf + ((rowbase + srow) * OBS + (size_t)t) * HID + part * 64);
#pragma unroll
            for (int i = 0; i < 8; ++i) dst[i] = src[i];   // 128 bytes per lane
        }
        __syncthreads();
    }
}

// ================= kernel 3: attention head + output (tiny, VALU) =================
__global__ void __launch_bounds__(128)
head_kernel(const bf16_t* __restrict__ H, const float* __restrict__ convW,
            const float* __restrict__ convb, const float* __restrict__ l1W,
            const float* __restrict__ l1b, const float* __restrict__ l2W,
            const float* __restrict__ l2b, const float* __restrict__ oW,
            const float* __restrict__ ob, float* __restrict__ out) {
    __shared__ float convW_s[FNUM * (OBS - 1)];
    __shared__ float conv_s[HID * FNUM];
    __shared__ float htt_s[HID];
    __shared__ float w_s[FNUM];
    __shared__ float alpha_s[HID];
    __shared__ float v_s[FNUM];
    __shared__ float nh_s[HID];

    const int    h = threadIdx.x;        // 0..127 (one hidden unit per thread)
    const size_t r = blockIdx.x;         // batch row
    for (int i = h; i < FNUM * (OBS - 1); i += 128) convW_s[i] = convW[i];

    const bf16_t* Hr = H + r * OBS * HID;
    float htt = (float)Hr[(OBS - 1) * HID + h];
    htt_s[h] = htt;

    float acc[FNUM];
#pragma unroll
    for (int f = 0; f < FNUM; ++f) acc[f] = convb[f];
    __syncthreads();

    for (int t = 0; t < OBS - 1; ++t) {
        float hv = fmaxf((float)Hr[t * HID + h], 0.0f);   // H = relu(hts[:-1])
#pragma unroll
        for (int f = 0; f < FNUM; ++f) acc[f] += hv * convW_s[f * (OBS - 1) + t];
    }
#pragma unroll
    for (int f = 0; f < FNUM; ++f) { acc[f] = fmaxf(acc[f], 0.0f); conv_s[h * FNUM + f] = acc[f]; }
    __syncthreads();

    if (h < FNUM) {                       // w = htt @ lin1_W.T + lin1_b
        float wv = l1b[h];
        for (int k = 0; k < HID; ++k) wv += htt_s[k] * l1W[h * HID + k];
        w_s[h] = wv;
    }
    __syncthreads();

    float aacc = 0.0f;                    // alpha = sigmoid(conv . w)
#pragma unroll
    for (int f = 0; f < FNUM; ++f) aacc += acc[f] * w_s[f];
    alpha_s[h] = fsig(aacc);
    __syncthreads();

    if (h < FNUM) {                       // v = alpha . conv
        float vv = 0.0f;
        for (int k = 0; k < HID; ++k) vv += alpha_s[k] * conv_s[k * FNUM + h];
        v_s[h] = vv;
    }
    __syncthreads();

    float nh = l2b[h];                    // new_ht = [htt, v] @ lin2_W.T + lin2_b
    for (int k = 0; k < HID; ++k) nh += htt_s[k] * l2W[h * (HID + FNUM) + k];
#pragma unroll
    for (int k = 0; k < FNUM; ++k) nh += v_s[k] * l2W[h * (HID + FNUM) + HID + k];
    nh_s[h] = nh;
    __syncthreads();

    if (h < HORIZON) {                    // y = new_ht @ out_W.T + out_b
        float yv = ob[h];
        for (int k = 0; k < HID; ++k) yv += nh_s[k] * oW[h * HID + k];
        out[r * HORIZON + h] = yv;
    }
}

// ================= launcher =================
extern "C" void kernel_launch(void* const* d_in, const int* in_sizes, int n_in,
                              void* d_out, int out_size, void* d_ws, size_t ws_size,
                              hipStream_t stream) {
    (void)in_sizes; (void)n_in; (void)out_size; (void)ws_size;
    const float* x     = (const float*)d_in[0];
    const float* Wh    = (const float*)d_in[1];
    const float* bh    = (const float*)d_in[2];
    const float* W_ih  = (const float*)d_in[3];
    const float* W_hh  = (const float*)d_in[4];
    const float* b_ih  = (const float*)d_in[5];
    const float* b_hh  = (const float*)d_in[6];
    const float* convW = (const float*)d_in[7];
    const float* convb = (const float*)d_in[8];
    const float* l1W   = (const float*)d_in[9];
    const float* l1b   = (const float*)d_in[10];
    const float* l2W   = (const float*)d_in[11];
    const float* l2b   = (const float*)d_in[12];
    const float* oW    = (const float*)d_in[13];
    const float* ob    = (const float*)d_in[14];
    float*  out = (float*)d_out;
    bf16_t* buf = (bf16_t*)d_ws;          // Be*24*128 bf16 activation buffer, reused per layer

    // 1) input projection -> bf16 buffer
    long total = (long)BE * OBS * HID;
    int  thr = 256;
    unsigned blk = (unsigned)((total + thr - 1) / thr);
    xc_kernel<<<blk, thr, 0, stream>>>(x, Wh, bh, buf, total);

    // 2) two LSTM layers, in place (layerwise; input-side of each layer is non-recurrent once
    //    previous layer's full sequence is materialized)
    size_t smem = (size_t)2 * 65536 * sizeof(bf16_t)   // W_ih + W_hh bf16 fragments (256KB)
                + (size_t)GATES * sizeof(float)        // fused bias
                + (size_t)4 * 2048 * sizeof(bf16_t);   // per-wave h staging tiles
    lstm_layer_kernel<<<BE / 64, 128, smem, stream>>>(W_ih, W_hh, b_ih, b_hh, buf);
    lstm_layer_kernel<<<BE / 64, 128, smem, stream>>>(W_ih + GATES * HID, W_hh + GATES * HID,
                                                      b_ih + GATES, b_hh + GATES, buf);

    // 3) attention head + output
    head_kernel<<<BE, 128, 0, stream>>>(buf, convW, convb, l1W, l1b, l2W, l2b, oW, ob, out);
}
